// MultiHeadAttention_47631187313085
// MI455X (gfx1250) — matrix operand order repaired and checked by reading, not compiled, not run on hardware
//
#include <hip/hip_runtime.h>
#include <math.h>

// Problem constants (match reference)
#define SEQ 4096
#define DM  1024
#define NH  16
#define DK  64
#define AV_WAVES 8

typedef __bf16 bf16_t;
typedef __attribute__((ext_vector_type(16))) __bf16 bf16x16;
typedef __attribute__((ext_vector_type(8)))  float  f32x8;

union FragU { bf16x16 v; uint4 q[2]; unsigned short u16[16]; };

static __device__ __forceinline__ f32x8 wmma_bf16(bf16x16 a, bf16x16 b, f32x8 c) {
  // 8 args: (neg_a, A, neg_b, B, c_mod, C, reuse_a, reuse_b)
  return __builtin_amdgcn_wmma_f32_16x16x32_bf16(false, a, false, b, (short)0, c,
                                                 false, false);
}

// --- A fragment: 16x32 bf16 tile, row-major source, per CDNA5 A layout.
// lane 0-15: elems 0..7 -> K=0..7, 8..15 -> K=16..23 ; lane 16-31: +8.
static __device__ __forceinline__ bf16x16 load_a_rm(const bf16_t* A, int lda, int lane) {
  const int m = lane & 15, kg = lane >> 4;
  const bf16_t* p = A + (size_t)m * lda + kg * 8;
  FragU f;
  f.q[0] = *(const uint4*)(p);
  f.q[1] = *(const uint4*)(p + 16);
  return f.v;
}

// --- B fragment, NT (B stored transposed: row n, contiguous along K).
// lane: n = lane&15, elems 0..7 -> K = (lane>>4)*8 + i, 8..15 -> K = 16 + (lane>>4)*8 + (i-8).
static __device__ __forceinline__ bf16x16 load_b_nt(const bf16_t* Bt, int ldbt, int lane) {
  const int n = lane & 15, kg = lane >> 4;
  const bf16_t* p = Bt + (size_t)n * ldbt + kg * 8;
  FragU f;
  f.q[0] = *(const uint4*)(p);
  f.q[1] = *(const uint4*)(p + 16);
  return f.v;
}

// =====================================================================
// bf16 NT GEMM:  C[M,N] = A[M,K] @ Bt[N,K]^T
// One wave per 64(M) x 32(N) tile; 8 f32 accumulators; K-loop step 32.
// All fragment loads are contiguous b128. STORET stores C transposed
// (C^T[N,M]) so downstream GEMMs can consume it as Bt.
// =====================================================================
template <bool F32OUT, bool STORET>
__global__ __launch_bounds__(32) void gemm_nt_kernel(
    const bf16_t* __restrict__ A, const bf16_t* __restrict__ Bt,
    void* __restrict__ Cout, int K, int lda, int ldbt, int ldc) {
  const int lane = threadIdx.x & 31;
  const int tn = blockIdx.x;   // N / 32 tiles
  const int tm = blockIdx.y;   // M / 64 tiles
  const bf16_t* Ab = A + (size_t)(tm * 64) * lda;
  const bf16_t* Bb = Bt + (size_t)(tn * 32) * ldbt;
  f32x8 acc[4][2] = {};
  for (int k0 = 0; k0 < K; k0 += 32) {
    if (k0 + 32 < K) {
      __builtin_prefetch(Ab + k0 + 32, 0, 1);
      __builtin_prefetch(Bb + k0 + 32, 0, 1);
    }
    const bf16x16 b0 = load_b_nt(Bb + k0, ldbt, lane);
    const bf16x16 b1 = load_b_nt(Bb + (size_t)16 * ldbt + k0, ldbt, lane);
#pragma unroll
    for (int j = 0; j < 4; ++j) {
      const bf16x16 a = load_a_rm(Ab + (size_t)(j * 16) * lda + k0, lda, lane);
      acc[j][0] = wmma_bf16(a, b0, acc[j][0]);
      acc[j][1] = wmma_bf16(a, b1, acc[j][1]);
    }
  }
  const int col = lane & 15, r0 = (lane >> 4) * 8;
#pragma unroll
  for (int j = 0; j < 4; ++j) {
#pragma unroll
    for (int n = 0; n < 2; ++n) {
#pragma unroll
      for (int r = 0; r < 8; ++r) {
        const size_t row = (size_t)(tm * 64 + j * 16 + r0 + r);
        const size_t gcol = (size_t)(tn * 32 + n * 16 + col);
        const size_t idx = STORET ? gcol * ldc + row : row * ldc + gcol;
        if (F32OUT)
          ((float*)Cout)[idx] = acc[j][n][r];
        else
          ((bf16_t*)Cout)[idx] = (bf16_t)acc[j][n][r];
      }
    }
  }
}

// =====================================================================
// Pass 1: per-(head, key-column s) online softmax stats over the QUERY axis.
// One wave per (head, 16-column s tile). K^T fragments are contiguous (NT).
// =====================================================================
__global__ __launch_bounds__(32) void attn_stats_kernel(
    const bf16_t* __restrict__ Qb, const bf16_t* __restrict__ Kb,
    float* __restrict__ Ml, float* __restrict__ Ll) {
  const int lane = threadIdx.x & 31;
  const int s0 = blockIdx.x * 16;
  const int h = blockIdx.y;
  const bf16_t* Qh = Qb + h * DK;
  const bf16_t* Kh = Kb + h * DK;
  // B fragments (K^T for this s tile) are loop-invariant: preload both k-steps.
  const bf16x16 bk0 = load_b_nt(Kh + (size_t)s0 * DM, DM, lane);
  const bf16x16 bk1 = load_b_nt(Kh + (size_t)s0 * DM + 32, DM, lane);
  float ml = -1e30f, sl = 0.0f;
  for (int q0 = 0; q0 < SEQ; q0 += 16) {
    const bf16x16 a0 = load_a_rm(Qh + (size_t)q0 * DM, DM, lane);
    const bf16x16 a1 = load_a_rm(Qh + (size_t)q0 * DM + 32, DM, lane);
    f32x8 c = {};
    c = wmma_bf16(a0, bk0, c);
    c = wmma_bf16(a1, bk1, c);
    float xs[8];
    float tm = -1e30f;
#pragma unroll
    for (int r = 0; r < 8; ++r) {
      const float x = c[r] * 0.125f;  // 1/sqrt(64)
      xs[r] = x;
      tm = fmaxf(tm, x);
    }
    const float mn = fmaxf(ml, tm);
    float s = sl * __expf(ml - mn);
#pragma unroll
    for (int r = 0; r < 8; ++r) s += __expf(xs[r] - mn);
    ml = mn;
    sl = s;
  }
  // Combine the two half-wave row groups that share a column (wave32 xor-16).
  const float mo = __shfl_xor(ml, 16, 32);
  const float so = __shfl_xor(sl, 16, 32);
  const float mc = fmaxf(ml, mo);
  const float sc = sl * __expf(ml - mc) + so * __expf(mo - mc);
  if (lane < 16) {
    Ml[h * SEQ + s0 + lane] = mc;
    Ll[h * SEQ + s0 + lane] = sc;
  }
}

// VpT[h*64+k][s] = V[s][h*64+k] / l[h][s]   (transposed so attn_av reads Bt)
__global__ void vscale_t_kernel(const bf16_t* __restrict__ Vb,
                                const float* __restrict__ Ll,
                                bf16_t* __restrict__ VpT) {
  const int i = blockIdx.x * blockDim.x + threadIdx.x;
  if (i >= SEQ * DM) return;
  const int s = i / DM;
  const int n = i % DM;       // n = h*64 + k
  const int h = n / DK;
  VpT[(size_t)n * SEQ + s] = (bf16_t)((float)Vb[i] / Ll[h * SEQ + s]);
}

// =====================================================================
// Pass 2: heads[q,k] = sum_s exp(score[q,s]-m[s]) * V'[s,k], concat layout.
// Block = 8 waves sharing one head; each wave owns a 16-row q tile and all
// 64 head-dim columns. P tiles go through per-wave LDS (C-layout->A-layout);
// V' fragments are contiguous NT loads from global (L2 resident).
// =====================================================================
__global__ __launch_bounds__(256) void attn_av_kernel(
    const bf16_t* __restrict__ Qb, const bf16_t* __restrict__ Kb,
    const bf16_t* __restrict__ VpT, const float* __restrict__ Ml,
    bf16_t* __restrict__ Hc) {
  __shared__ __align__(16) bf16_t Pt[AV_WAVES][16 * 32];  // per-wave P tile
  const int tid = threadIdx.x;
  const int lane = tid & 31;
  const int wave = tid >> 5;
  const int h = blockIdx.y;
  const int q0 = (blockIdx.x * AV_WAVES + wave) * 16;
  const bf16_t* Qh = Qb + h * DK;
  const bf16_t* Kh = Kb + h * DK;
  const bf16_t* Vh = VpT + (size_t)(h * DK) * SEQ;
  const bf16x16 aq0 = load_a_rm(Qh + (size_t)q0 * DM, DM, lane);
  const bf16x16 aq1 = load_a_rm(Qh + (size_t)q0 * DM + 32, DM, lane);
  f32x8 acc[4] = {};
  for (int s0 = 0; s0 < SEQ; s0 += 32) {
    // Build P = exp(score - m[s]) for this wave's 16x32 tile, stage to LDS.
#pragma unroll
    for (int sub = 0; sub < 2; ++sub) {
      const int sc0 = s0 + sub * 16;
      const bf16x16 b0 = load_b_nt(Kh + (size_t)sc0 * DM, DM, lane);
      const bf16x16 b1 = load_b_nt(Kh + (size_t)sc0 * DM + 32, DM, lane);
      f32x8 c = {};
      c = wmma_bf16(aq0, b0, c);
      c = wmma_bf16(aq1, b1, c);
      const float mlane = Ml[h * SEQ + sc0 + (lane & 15)];
      const int col = lane & 15, r0 = (lane >> 4) * 8;
#pragma unroll
      for (int r = 0; r < 8; ++r) {
        const float p = __expf(c[r] * 0.125f - mlane);
        Pt[wave][(r0 + r) * 32 + sub * 16 + col] = (bf16_t)p;
      }
    }
    __syncthreads();  // P stores visible across lanes before A-layout reload
    const bf16x16 a2 = load_a_rm(&Pt[wave][0], 32, lane);
#pragma unroll
    for (int j = 0; j < 4; ++j) {
      const bf16x16 b2 = load_b_nt(Vh + (size_t)(j * 16) * SEQ + s0, SEQ, lane);
      acc[j] = wmma_bf16(a2, b2, acc[j]);
    }
  }
  const int col = lane & 15, r0 = (lane >> 4) * 8;
#pragma unroll
  for (int j = 0; j < 4; ++j)
#pragma unroll
    for (int r = 0; r < 8; ++r)
      Hc[(size_t)(q0 + r0 + r) * DM + h * DK + j * 16 + col] = (bf16_t)acc[j][r];
}

__global__ void cast_f32_bf16_kernel(const float* __restrict__ src,
                                     bf16_t* __restrict__ dst, int n) {
  const int i = blockIdx.x * blockDim.x + threadIdx.x;
  if (i < n) dst[i] = (bf16_t)src[i];
}

// Batched cast + transpose: src fp32 [B][R][C] -> dst bf16 [B][C][R]
__global__ void cast_transpose_kernel(const float* __restrict__ src,
                                      bf16_t* __restrict__ dst, int R, int C,
                                      int total) {
  const int i = blockIdx.x * blockDim.x + threadIdx.x;
  if (i >= total) return;
  const int c = i % C;
  const int r = (i / C) % R;
  const int b = i / (R * C);
  dst[((size_t)b * C + c) * R + r] = (bf16_t)src[i];
}

// =====================================================================
extern "C" void kernel_launch(void* const* d_in, const int* in_sizes, int n_in,
                              void* d_out, int out_size, void* d_ws,
                              size_t ws_size, hipStream_t stream) {
  (void)in_sizes; (void)n_in; (void)out_size; (void)ws_size;
  const float* XQ = (const float*)d_in[0];
  const float* XK = (const float*)d_in[1];
  const float* XV = (const float*)d_in[2];
  const float* WQc = (const float*)d_in[3];
  const float* WKc = (const float*)d_in[4];
  const float* WVc = (const float*)d_in[5];
  const float* WQh = (const float*)d_in[6];
  const float* WKh = (const float*)d_in[7];
  const float* WVh = (const float*)d_in[8];
  const float* WO = (const float*)d_in[9];
  float* out = (float*)d_out;

  char* ws = (char*)d_ws;
  size_t off = 0;
  auto salloc = [&](size_t nelem, size_t esz) -> void* {
    void* p = ws + off;
    off += (nelem * esz + 255) & ~(size_t)255;
    return p;
  };
  bf16_t* XQb = (bf16_t*)salloc(SEQ * DM, 2);
  bf16_t* XKb = (bf16_t*)salloc(SEQ * DM, 2);
  bf16_t* XVb = (bf16_t*)salloc(SEQ * DM, 2);
  bf16_t* WQcb = (bf16_t*)salloc(DM * DM, 2);
  bf16_t* WKcb = (bf16_t*)salloc(DM * DM, 2);
  bf16_t* WVcb = (bf16_t*)salloc(DM * DM, 2);
  bf16_t* WQhT = (bf16_t*)salloc(DM * DM, 2);  // [h*64+c][e] flat = Bt
  bf16_t* WKhT = (bf16_t*)salloc(DM * DM, 2);
  bf16_t* WVhT = (bf16_t*)salloc(DM * DM, 2);
  bf16_t* WOT = (bf16_t*)salloc(DM * DM, 2);   // WO^T
  bf16_t* WeQT = (bf16_t*)salloc(DM * DM, 2);  // W_eff^T (stored by STORET)
  bf16_t* WeKT = (bf16_t*)salloc(DM * DM, 2);
  bf16_t* WeVT = (bf16_t*)salloc(DM * DM, 2);
  bf16_t* Qb = (bf16_t*)salloc(SEQ * DM, 2);
  bf16_t* Kb = (bf16_t*)salloc(SEQ * DM, 2);
  bf16_t* Vb = (bf16_t*)salloc(SEQ * DM, 2);
  bf16_t* VpT = (bf16_t*)salloc(SEQ * DM, 2);  // [h*64+k][s]
  bf16_t* Hc = (bf16_t*)salloc(SEQ * DM, 2);
  float* Mlp = (float*)salloc(NH * SEQ, 4);
  float* Llp = (float*)salloc(NH * SEQ, 4);

  // 1) fp32 -> bf16 casts (plain) + transposing casts for Bt operands
  {
    struct Job { const float* s; bf16_t* d; int n; };
    const Job jobs[6] = {{XQ, XQb, SEQ * DM},  {XK, XKb, SEQ * DM},
                         {XV, XVb, SEQ * DM},  {WQc, WQcb, DM * DM},
                         {WKc, WKcb, DM * DM}, {WVc, WVcb, DM * DM}};
    for (int i = 0; i < 6; ++i)
      cast_f32_bf16_kernel<<<(jobs[i].n + 255) / 256, 256, 0, stream>>>(
          jobs[i].s, jobs[i].d, jobs[i].n);
    struct TJob { const float* s; bf16_t* d; int R, C; };
    const TJob tj[4] = {{WQh, WQhT, DM, DK},   // batch NH folds into flat idx
                        {WKh, WKhT, DM, DK},
                        {WVh, WVhT, DM, DK},
                        {WO, WOT, DM, DM}};
    for (int i = 0; i < 4; ++i) {
      const int total = DM * DM;  // all four are 1M elements total
      cast_transpose_kernel<<<(total + 255) / 256, 256, 0, stream>>>(
          tj[i].s, tj[i].d, tj[i].R, tj[i].C, total);
    }
  }
  // 2) Effective per-head weights, stored transposed:
  //    WeT[n][d] = sum_e Wc[d][e] * WhT[n][e]
  {
    dim3 g(DM / 32, DM / 64);
    gemm_nt_kernel<false, true><<<g, 32, 0, stream>>>(WQcb, WQhT, WeQT, DM, DM,
                                                      DM, DM);
    gemm_nt_kernel<false, true><<<g, 32, 0, stream>>>(WKcb, WKhT, WeKT, DM, DM,
                                                      DM, DM);
    gemm_nt_kernel<false, true><<<g, 32, 0, stream>>>(WVcb, WVhT, WeVT, DM, DM,
                                                      DM, DM);
  }
  // 3) Projections: Q/K/V[s][h*64+k] = X @ We  (B consumed transposed)
  {
    dim3 g(DM / 32, SEQ / 64);
    gemm_nt_kernel<false, false><<<g, 32, 0, stream>>>(XQb, WeQT, Qb, DM, DM,
                                                       DM, DM);
    gemm_nt_kernel<false, false><<<g, 32, 0, stream>>>(XKb, WeKT, Kb, DM, DM,
                                                       DM, DM);
    gemm_nt_kernel<false, false><<<g, 32, 0, stream>>>(XVb, WeVT, Vb, DM, DM,
                                                       DM, DM);
  }
  // 4) Column (query-axis) softmax stats per (head, key)
  {
    dim3 g(SEQ / 16, NH);
    attn_stats_kernel<<<g, 32, 0, stream>>>(Qb, Kb, Mlp, Llp);
  }
  // 5) Fold 1/l into V, store transposed [h*64+k][s]
  vscale_t_kernel<<<(SEQ * DM + 255) / 256, 256, 0, stream>>>(Vb, Llp, VpT);
  // 6) heads = exp(score - m) @ V'
  {
    dim3 g(SEQ / (16 * AV_WAVES), NH);
    attn_av_kernel<<<g, 256, 0, stream>>>(Qb, Kb, VpT, Mlp, Hc);
  }
  // 7) out = concat(heads) @ WO  (fp32 output)
  {
    dim3 g(DM / 32, SEQ / 64);
    gemm_nt_kernel<true, false><<<g, 32, 0, stream>>>(Hc, WOT, out, DM, DM, DM,
                                                      DM);
  }
}
